// PatchLossOptimized_43052752175786
// MI455X (gfx1250) — matrix operand order repaired and checked by reading, not compiled, not run on hardware
//
#include <hip/hip_runtime.h>
#include <hip/hip_bf16.h>

// Problem constants (from reference): B=256, T=196, D=768, K=4
#define T_DIM 196
#define D_DIM 768
#define NT    13                      // ceil(196/16) tiles along T
#define NU2   7                       // ceil(13/2) u-tile pairs
#define PAIRS2 (NT * NU2)             // 91 wave-jobs per batch (each = 16x32 of G)
#define WAVES 8                       // waves per workgroup (256 threads, wave32)
#define GROUPS_PER_B ((PAIRS2 + WAVES - 1) / WAVES)  // 12

typedef __attribute__((ext_vector_type(16))) __bf16         v16bf;
typedef __attribute__((ext_vector_type(8)))  float          v8f;
typedef __attribute__((ext_vector_type(8)))  unsigned short u16x8;

union ABFrag { u16x8 h[2]; v16bf v; };

__device__ __forceinline__ unsigned short f32_to_bf16_rne(float f) {
  unsigned u = __builtin_bit_cast(unsigned, f);
  unsigned r = 0x7FFFu + ((u >> 16) & 1u);
  return (unsigned short)((u + r) >> 16);
}

// ---------------------------------------------------------------------------
// one-time: inv2s2[k] = 1/(2*sigma[k]^2)  (hoists the IEEE-div ladder out of
// the 24.6k-wave tile kernel)
// ---------------------------------------------------------------------------
__global__ void sigma_prep_kernel(const float* __restrict__ sigma,
                                  float* __restrict__ is4) {
  const int k = threadIdx.x;
  if (k < 4) {
    float s = sigma[k];
    is4[k] = 1.0f / (2.0f * s * s);
  }
}

// ---------------------------------------------------------------------------
// prep: per (b,t) row -> z2[b*T+t] = sum_d z^2 ; optionally bf16 copy of z
// grid = B*T blocks, 256 threads
// ---------------------------------------------------------------------------
__global__ __launch_bounds__(256)
void prep_kernel(const float* __restrict__ z,
                 unsigned short* __restrict__ zb,   // used iff write_bf16
                 float* __restrict__ z2,
                 int write_bf16) {
  const int row = blockIdx.x;                 // b*T_DIM + t
  const int tid = threadIdx.x;
  const float* zr = z + (size_t)row * D_DIM;
  float ss = 0.f;
  if (write_bf16) {
    unsigned short* zbr = zb + (size_t)row * D_DIM;
    for (int i = tid; i < D_DIM; i += 256) {
      float v = zr[i];
      zbr[i] = f32_to_bf16_rne(v);
      ss += v * v;
    }
  } else {
    for (int i = tid; i < D_DIM; i += 256) {
      float v = zr[i];
      ss += v * v;
    }
  }
  for (int off = 16; off >= 1; off >>= 1) ss += __shfl_xor(ss, off, 32);
  __shared__ float sred[WAVES];
  if ((tid & 31) == 0) sred[tid >> 5] = ss;
  __syncthreads();
  if (tid == 0) {
    float s = 0.f;
#pragma unroll
    for (int i = 0; i < WAVES; ++i) s += sred[i];
    z2[row] = s;
  }
}

// Convert 8 consecutive f32 -> bf16 slots [base, base+8) of a v16bf fragment.
__device__ __forceinline__ void cvt8(const float* __restrict__ p, v16bf& v, int base) {
  float4 f0 = *(const float4*)(p);
  float4 f1 = *(const float4*)(p + 4);
  v[base + 0] = (__bf16)f0.x; v[base + 1] = (__bf16)f0.y;
  v[base + 2] = (__bf16)f0.z; v[base + 3] = (__bf16)f0.w;
  v[base + 4] = (__bf16)f1.x; v[base + 5] = (__bf16)f1.y;
  v[base + 6] = (__bf16)f1.z; v[base + 7] = (__bf16)f1.w;
}

// ---------------------------------------------------------------------------
// main: one wave per 16(t) x 32(u) block of one batch: two 16x16 WMMA
// accumulators sharing one A fragment (3 b128 loads per v_wmma instead of 4).
//   G via v_wmma_f32_16x16x32_bf16 over D=768 (24 steps x 2 wmma),
//   then fused RBF weight + reduction of w*(z2_t + z2_u - 2G).
// grid = B*GROUPS_PER_B blocks, 256 threads (8 waves)
// ---------------------------------------------------------------------------
template <bool USE_BF16_WS>
__global__ __launch_bounds__(256)
void tile_kernel(const float* __restrict__ gt_dT,
                 const float* __restrict__ is4,       // 1/(2*sigma^2), 4 floats
                 const float* __restrict__ z,         // f32 path
                 const unsigned short* __restrict__ zb, // bf16 path
                 const float* __restrict__ z2,
                 float* __restrict__ partials) {
  __shared__ float sred[WAVES];
  const int g    = blockIdx.x;
  const int b    = g / GROUPS_PER_B;
  const int gi   = g % GROUPS_PER_B;
  const int wave = threadIdx.x >> 5;
  const int lane = threadIdx.x & 31;
  const int tile = gi * WAVES + wave;

  float acc = 0.f;
  if (tile < PAIRS2) {                        // uniform per wave -> EXEC all-1s
    const int ti = tile / NU2, uj = tile % NU2;
    const int tbase  = ti * 16;
    const int ubase0 = uj * 32;
    const int ubase1 = ubase0 + 16;           // may exceed T (tiles clamped+masked)

    // A fragment (16x32 bf16): lane L -> row M = tbase + L%16, Kbase = (L/16)*16,
    //   VGPR0..7: K = Kbase+0..15 (16 consecutive K per lane)
    // B fragment (32x16 bf16): lane L -> col N = L%16, Kbase = (L/16)*16,
    //   VGPR j: K = Kbase+2j, Kbase+2j+1  (16 consecutive K per lane)
    const int arow  = tbase  + (lane & 15);
    const int brow0 = ubase0 + (lane & 15);
    const int brow1 = ubase1 + (lane & 15);
    const int arc  = (arow  < T_DIM) ? arow  : (T_DIM - 1);  // clamp; masked later
    const int brc0 = (brow0 < T_DIM) ? brow0 : (T_DIM - 1);
    const int brc1 = (brow1 < T_DIM) ? brow1 : (T_DIM - 1);
    const int akb  = (lane >> 4) * 16;
    const int bkb  = (lane >> 4) * 16;

    v8f c0 = {}, c1 = {};
    if (USE_BF16_WS) {
      const __bf16* base = (const __bf16*)zb + (size_t)b * T_DIM * D_DIM;
      const __bf16* ap  = base + (size_t)arc  * D_DIM + akb;
      const __bf16* bp0 = base + (size_t)brc0 * D_DIM + bkb;
      const __bf16* bp1 = base + (size_t)brc1 * D_DIM + bkb;
#pragma unroll 2
      for (int k0 = 0; k0 < D_DIM; k0 += 32) {
        ABFrag a, b0, b1;
        a.h[0]  = *(const u16x8*)(ap  + k0);
        a.h[1]  = *(const u16x8*)(ap  + k0 + 8);
        b0.h[0] = *(const u16x8*)(bp0 + k0);
        b0.h[1] = *(const u16x8*)(bp0 + k0 + 8);
        b1.h[0] = *(const u16x8*)(bp1 + k0);
        b1.h[1] = *(const u16x8*)(bp1 + k0 + 8);
        c0 = __builtin_amdgcn_wmma_f32_16x16x32_bf16(false, a.v, false, b0.v,
                                                     (short)0, c0, false, false);
        c1 = __builtin_amdgcn_wmma_f32_16x16x32_bf16(false, a.v, false, b1.v,
                                                     (short)0, c1, false, false);
      }
    } else {
      const float* base = z + (size_t)b * T_DIM * D_DIM;
      const float* ap  = base + (size_t)arc  * D_DIM + akb;
      const float* bp0 = base + (size_t)brc0 * D_DIM + bkb;
      const float* bp1 = base + (size_t)brc1 * D_DIM + bkb;
#pragma unroll 2
      for (int k0 = 0; k0 < D_DIM; k0 += 32) {
        v16bf av, bv0, bv1;
        cvt8(ap  + k0,      av,  0);
        cvt8(ap  + k0 + 8,  av,  8);
        cvt8(bp0 + k0,      bv0, 0);
        cvt8(bp0 + k0 + 8,  bv0, 8);
        cvt8(bp1 + k0,      bv1, 0);
        cvt8(bp1 + k0 + 8,  bv1, 8);
        c0 = __builtin_amdgcn_wmma_f32_16x16x32_bf16(false, av, false, bv0,
                                                     (short)0, c0, false, false);
        c1 = __builtin_amdgcn_wmma_f32_16x16x32_bf16(false, av, false, bv1,
                                                     (short)0, c1, false, false);
      }
    }

    // Fused RBF weight + reduction. C layout: VGPR r, lane L ->
    //   M = r + 8*(L/16), N = L%16
    const float is0 = is4[0], is1 = is4[1], is2 = is4[2], is3 = is4[3];
    const float* z2b = z2 + (size_t)b * T_DIM;
#pragma unroll
    for (int half = 0; half < 2; ++half) {
      const v8f& c = half ? c1 : c0;
      const int u = (half ? ubase1 : ubase0) + (lane & 15);
      const float z2u = z2b[(u < T_DIM) ? u : (T_DIM - 1)];
#pragma unroll
      for (int r = 0; r < 8; ++r) {
        const int t = tbase + r + ((lane >> 4) << 3);
        if (t < T_DIM && u < T_DIM) {
          const float4 gv = *(const float4*)(gt_dT +
              ((((size_t)b * T_DIM + t) * T_DIM + u) << 2));
          float kv = gv.x * gv.x * is0 + gv.y * gv.y * is1 +
                     gv.z * gv.z * is2 + gv.w * gv.w * is3;
          acc += __expf(-kv) * (z2b[t] + z2u - 2.0f * c[r]);
        }
      }
    }
  }

  for (int off = 16; off >= 1; off >>= 1) acc += __shfl_xor(acc, off, 32);
  if (lane == 0) sred[wave] = acc;
  __syncthreads();
  if (threadIdx.x == 0) {
    float s = 0.f;
#pragma unroll
    for (int i = 0; i < WAVES; ++i) s += sred[i];
    partials[g] = s;
  }
}

// ---------------------------------------------------------------------------
// final: deterministic sum of per-workgroup partials, scaled by 1/(B*T*T)
// ---------------------------------------------------------------------------
__global__ __launch_bounds__(256)
void final_reduce(const float* __restrict__ partials, int n,
                  float* __restrict__ out, float scale) {
  float s = 0.f;
  for (int i = threadIdx.x; i < n; i += 256) s += partials[i];
  for (int off = 16; off >= 1; off >>= 1) s += __shfl_xor(s, off, 32);
  __shared__ float sred[WAVES];
  if ((threadIdx.x & 31) == 0) sred[threadIdx.x >> 5] = s;
  __syncthreads();
  if (threadIdx.x == 0) {
    float tot = 0.f;
#pragma unroll
    for (int i = 0; i < WAVES; ++i) tot += sred[i];
    out[0] = tot * scale;
  }
}

extern "C" void kernel_launch(void* const* d_in, const int* in_sizes, int n_in,
                              void* d_out, int out_size, void* d_ws, size_t ws_size,
                              hipStream_t stream) {
  const float* z     = (const float*)d_in[0];   // [B,T,D] f32
  const float* gt_dT = (const float*)d_in[1];   // [B,T,T,4] f32
  const float* sigma = (const float*)d_in[2];   // [4] f32
  float* out = (float*)d_out;

  const int B = in_sizes[0] / (T_DIM * D_DIM);
  const int ngroups = B * GROUPS_PER_B;

  // Workspace: [ z2: B*T f32 ][ partials: ngroups f32 ][ is4: 4 f32 ][ zb: B*T*D bf16 ]
  float* z2       = (float*)d_ws;
  float* partials = z2 + (size_t)B * T_DIM;
  float* is4      = partials + ngroups;
  size_t head_bytes = ((size_t)B * T_DIM + (size_t)ngroups + 4) * sizeof(float);
  head_bytes = (head_bytes + 255) & ~(size_t)255;                 // align zb
  unsigned short* zb = (unsigned short*)((char*)d_ws + head_bytes);
  size_t need_bf16 = head_bytes + (size_t)B * T_DIM * D_DIM * sizeof(unsigned short);
  const bool use_ws = (ws_size >= need_bf16);

  sigma_prep_kernel<<<1, 32, 0, stream>>>(sigma, is4);
  prep_kernel<<<B * T_DIM, 256, 0, stream>>>(z, zb, z2, use_ws ? 1 : 0);
  if (use_ws) {
    tile_kernel<true><<<ngroups, 256, 0, stream>>>(gt_dT, is4, z, zb, z2, partials);
  } else {
    tile_kernel<false><<<ngroups, 256, 0, stream>>>(gt_dT, is4, z, zb, z2, partials);
  }
  final_reduce<<<1, 256, 0, stream>>>(partials, ngroups, out,
                                      1.0f / ((float)B * (float)T_DIM * (float)T_DIM));
  (void)n_in; (void)out_size;
}
